// MultiHeadAttention_19885698580865
// MI455X (gfx1250) — compile-verified, matrix operand order checked by reading, not run on hardware
//
#include <hip/hip_runtime.h>
#include <hip/hip_bf16.h>

// ---------------------------------------------------------------------------
// MHA for MI455X (gfx1250): WMMA f16 GEMMs + TDM (tensor_load_to_lds) staging.
// ---------------------------------------------------------------------------

#define BATCH  2
#define SEQ    2048
#define HEADS  16
#define HDIM   64
#define DMODEL 1024

typedef __attribute__((ext_vector_type(16))) _Float16 v16h;
typedef __attribute__((ext_vector_type(8)))  _Float16 v8h;
typedef __attribute__((ext_vector_type(8)))  float    v8f;
typedef __attribute__((ext_vector_type(4)))  unsigned v4u;
typedef __attribute__((ext_vector_type(4)))  int      v4i;
typedef __attribute__((ext_vector_type(8)))  int      v8i;

#ifndef __has_builtin
#define __has_builtin(x) 0
#endif
#if __has_builtin(__builtin_amdgcn_tensor_load_to_lds)
#define HAVE_TDM 1
#else
#define HAVE_TDM 0
#endif

// ---------------------------------------------------------------------------
// 16-bit A/B tile load matching CDNA5 WMMA 16x32 VGPR layout:
// lane L (0..15): row/col = L, K = {0..7, 16..23}; lane L+16: K = {8..15, 24..31}.
// ---------------------------------------------------------------------------
__device__ __forceinline__ v16h lds_tile_16x32(const _Float16* base, int stride) {
    int lane = threadIdx.x & 31;
    int r    = lane & 15;
    int kb   = (lane >> 4) << 3;            // 0 or 8
    const _Float16* p = base + r * stride + kb;
    v8h lo = *(const v8h*)(p);              // K = kb .. kb+7
    v8h hi = *(const v8h*)(p + 16);         // K = kb+16 .. kb+23
    v16h a;
#pragma unroll
    for (int i = 0; i < 8; ++i) { a[i] = lo[i]; a[i + 8] = hi[i]; }
    return a;
}

__device__ __forceinline__ v8f wmma_f16(v16h a, v16h b, v8f c) {
    return __builtin_amdgcn_wmma_f32_16x16x32_f16(false, a, false, b, (short)0, c,
                                                  false, false);
}

__device__ __forceinline__ void wait_tensorcnt0() {
#if __has_builtin(__builtin_amdgcn_s_wait_tensorcnt)
    __builtin_amdgcn_s_wait_tensorcnt(0);
#else
    asm volatile("s_wait_tensorcnt 0x0" ::: "memory");
#endif
}

// Low 32 bits of a generic pointer to __shared__ == LDS byte address.
__device__ __forceinline__ unsigned lds_off(const void* p) {
    return (unsigned)(size_t)p;
}

#if HAVE_TDM
// TDM 2-D f16 tile load: rows x 64 elements, contiguous rows in memory,
// LDS rows padded by 16B per 128B (-> 72-half row stride, 16B aligned).
// Tensor dims == tile dims (tile fully in-bounds by construction).
__device__ __forceinline__ void tdm_load_tile_f16(unsigned lds_byte,
                                                  const _Float16* gsrc,
                                                  unsigned rows) {
    unsigned long long ga = (unsigned long long)(size_t)gsrc;
    const unsigned dim0 = HDIM;      // 64 elements per row
    v4u g0 = {1u,                                   // count=1, no gather
              lds_byte,                             // lds_addr
              (unsigned)ga,                         // global_addr[31:0]
              (unsigned)((ga >> 32) & 0x01FFFFFFu) | (2u << 30)}; // addr[56:32]|type=2
    v8i g1;
    g1[0] = (int)((1u << 16)     // data_size = 2 bytes
                  | (1u << 20)   // pad_enable
                  | (4u << 22)   // pad_interval: 32 DWORDs = 128 B
                  | (3u << 25)); // pad_amount:   4 DWORDs = 16 B
    g1[1] = (int)((dim0 & 0xFFFFu) << 16);                        // tensor_dim0 lo
    g1[2] = (int)(((dim0 >> 16) & 0xFFFFu) | ((rows & 0xFFFFu) << 16)); // dim0 hi|dim1 lo
    g1[3] = (int)(((rows >> 16) & 0xFFFFu) | ((dim0 & 0xFFFFu) << 16)); // dim1 hi|tile0
    g1[4] = (int)(rows & 0xFFFFu);                                // tile_dim1 (tile2=0)
    g1[5] = (int)dim0;                                            // dim0_stride lo32
    g1[6] = 0;                                                    // stride hi | d1s lo
    g1[7] = 0;
    v4i z4 = {0, 0, 0, 0};
#if __clang_major__ >= 23
    v8i z8 = {0, 0, 0, 0, 0, 0, 0, 0};
    __builtin_amdgcn_tensor_load_to_lds(g0, g1, z4, z4, z8, 0);
#else
    __builtin_amdgcn_tensor_load_to_lds(g0, g1, z4, z4, 0);
#endif
}
#endif

// ---------------------------------------------------------------------------
// GEMM: out[M x 1024] = A[M x 1024] * W[1024 x 1024] + bias.
// Block = 256 threads (8 waves); tile 128(M) x 64(N); K step 32.
// Each wave owns a 16x64 strip: 1 A-load + 4 B-loads + 4 WMMAs per K slab.
// ---------------------------------------------------------------------------
template <bool A_F16, bool SPLIT_HEADS>
__global__ __launch_bounds__(256) void gemm_proj_kernel(
    const void* __restrict__ Aptr, const float* __restrict__ W,
    const float* __restrict__ bias, void* __restrict__ outp) {
    constexpr int Dk = DMODEL, Dn = DMODEL;

    __shared__ _Float16 As[128][40];  // [m][k], 80B rows (16B aligned)
    __shared__ _Float16 Bs[64][40];   // [n][k] = W^T tile

    const int t    = threadIdx.x;
    const int w    = t >> 5;
    const int lane = t & 31;
    const int half = lane >> 4;
    const int nl   = lane & 15;
    const int m0   = blockIdx.y * 128;
    const int n0   = blockIdx.x * 64;

    v8f acc[4];
#pragma unroll
    for (int nt = 0; nt < 4; ++nt)
#pragma unroll
        for (int r = 0; r < 8; ++r) acc[nt][r] = 0.f;

    for (int k0 = 0; k0 < Dk; k0 += 32) {
        // ---- stage A tile 128x32 (16 elements per thread) ----
        {
            int row = t >> 1;              // 0..127
            int kg  = (t & 1) * 16;        // 0,16
            if constexpr (A_F16) {
                const _Float16* Ag =
                    (const _Float16*)Aptr + (size_t)(m0 + row) * Dk + k0 + kg;
                *(v8h*)&As[row][kg]     = *(const v8h*)Ag;
                *(v8h*)&As[row][kg + 8] = *(const v8h*)(Ag + 8);
            } else {
                const float* Ag = (const float*)Aptr + (size_t)(m0 + row) * Dk + k0 + kg;
                _Float16* dst = &As[row][kg];
#pragma unroll
                for (int q4 = 0; q4 < 4; ++q4) {
                    float4 x = *(const float4*)(Ag + q4 * 4);
                    dst[q4 * 4 + 0] = (_Float16)x.x;
                    dst[q4 * 4 + 1] = (_Float16)x.y;
                    dst[q4 * 4 + 2] = (_Float16)x.z;
                    dst[q4 * 4 + 3] = (_Float16)x.w;
                }
                if (k0 + 32 < Dk) __builtin_prefetch(Ag + 32, 0, 1);
            }
        }
        // ---- stage B tile: W[k0+kk][n0+nn] -> Bs[nn][kk] ----
        {
            int kk = t >> 3;               // 0..31
            int ng = (t & 7) * 8;          // 0..56
            const float* Wg = W + (size_t)(k0 + kk) * Dn + n0 + ng;
            float4 x0 = *(const float4*)(Wg);
            float4 x1 = *(const float4*)(Wg + 4);
            Bs[ng + 0][kk] = (_Float16)x0.x; Bs[ng + 1][kk] = (_Float16)x0.y;
            Bs[ng + 2][kk] = (_Float16)x0.z; Bs[ng + 3][kk] = (_Float16)x0.w;
            Bs[ng + 4][kk] = (_Float16)x1.x; Bs[ng + 5][kk] = (_Float16)x1.y;
            Bs[ng + 6][kk] = (_Float16)x1.z; Bs[ng + 7][kk] = (_Float16)x1.w;
            if (k0 + 32 < Dk) __builtin_prefetch(Wg + 32 * Dn, 0, 1);
        }
        __syncthreads();

        v16h a = lds_tile_16x32(&As[w * 16][0], 40);
#pragma unroll
        for (int nt = 0; nt < 4; ++nt) {
            v16h b = lds_tile_16x32(&Bs[nt * 16][0], 40);
            acc[nt] = wmma_f16(a, b, acc[nt]);
        }
        __syncthreads();
    }

    // ---- epilogue: (reg r, lane L) -> row r+8*(L>=16), col L&15 ----
#pragma unroll
    for (int r = 0; r < 8; ++r) {
        int m  = m0 + w * 16 + r + 8 * half;
        int bb = m / SEQ, s = m % SEQ;
#pragma unroll
        for (int nt = 0; nt < 4; ++nt) {
            int n = n0 + nt * 16 + nl;
            float val = acc[nt][r] + bias[n];
            if constexpr (SPLIT_HEADS) {
                _Float16* o = (_Float16*)outp;
                o[(((size_t)bb * HEADS + (n >> 6)) * SEQ + s) * HDIM + (n & 63)] =
                    (_Float16)val;
            } else {
                float* o = (float*)outp;
                o[(size_t)m * Dn + n] = val;
            }
        }
    }
}

// ---------------------------------------------------------------------------
// Flash attention per (b,h): 8 waves / 128 query rows per block, 64-key tiles,
// online softmax. Q/K tiles staged by the Tensor Data Mover (pad -> 72-h rows).
// ---------------------------------------------------------------------------
__global__ __launch_bounds__(256) void flash_attn_kernel(
    const _Float16* __restrict__ qh, const _Float16* __restrict__ kh,
    const _Float16* __restrict__ vh, const float* __restrict__ mask,
    _Float16* __restrict__ ctx) {
    const int nqt = SEQ / 128;
    const int bh  = blockIdx.x / nqt;
    const int qt  = blockIdx.x % nqt;
    const int bb  = bh / HEADS;
    const int h   = bh % HEADS;
    const int q0  = qt * 128;

    const _Float16* Qg = qh + ((size_t)bh * SEQ + q0) * HDIM;
    const _Float16* Kg = kh + (size_t)bh * SEQ * HDIM;
    const _Float16* Vg = vh + (size_t)bh * SEQ * HDIM;

    __shared__ _Float16 Qs[128][72];      // [q][hd]
    __shared__ _Float16 Ks[64][72];       // [keypos][hd]  (B for QK^T)
    __shared__ _Float16 Vt[64][72];       // [hd][keypos]  (B for P*V)
    __shared__ _Float16 Ps[8][16][72];    // per-wave P staging (C -> A layout)
    __shared__ float    mk[64];

    const int t    = threadIdx.x;
    const int w    = t >> 5;
    const int lane = t & 31;
    const int half = lane >> 4;
    const int nl   = lane & 15;

    // ---- load Q tile once (128 rows x 64 halves) ----
#if HAVE_TDM
    if (w == 0) tdm_load_tile_f16(lds_off(&Qs[0][0]), Qg, 128);
#else
    {
        int row = t >> 1, cg = (t & 1) * 32;
#pragma unroll
        for (int j = 0; j < 4; ++j)
            *(v8h*)&Qs[row][cg + j * 8] =
                *(const v8h*)(Qg + (size_t)row * HDIM + cg + j * 8);
    }
#endif

    float mrow[8], lrow[8];
    v8f o[4];
#pragma unroll
    for (int r = 0; r < 8; ++r) { mrow[r] = -1e30f; lrow[r] = 0.f; }
#pragma unroll
    for (int nt = 0; nt < 4; ++nt)
#pragma unroll
        for (int r = 0; r < 8; ++r) o[nt][r] = 0.f;

    for (int kt = 0; kt < SEQ; kt += 64) {
        // ---- stage K tile (async via TDM) ----
#if HAVE_TDM
        if (w == 0) tdm_load_tile_f16(lds_off(&Ks[0][0]), Kg + (size_t)kt * HDIM, 64);
#else
        {
            int row = t >> 2, cg = (t & 3) * 16;
            *(v8h*)&Ks[row][cg]     = *(const v8h*)(Kg + (size_t)(kt + row) * HDIM + cg);
            *(v8h*)&Ks[row][cg + 8] = *(const v8h*)(Kg + (size_t)(kt + row) * HDIM + cg + 8);
        }
#endif
        // ---- stage V tile transposed: Vt[hd][keypos] ----
        for (int idx = t; idx < 64 * 8; idx += 256) {
            int kp = idx >> 3, hg = (idx & 7) * 8;
            v8h x = *(const v8h*)(Vg + (size_t)(kt + kp) * HDIM + hg);
#pragma unroll
            for (int j = 0; j < 8; ++j) Vt[hg + j][kp] = x[j];
        }
        if (t < 64) mk[t] = mask[(size_t)bb * SEQ + kt + t];
#if HAVE_TDM
        if (w == 0) wait_tensorcnt0();
#endif
        __syncthreads();

        // ---- S = Q K^T : wave w owns rows [w*16, w*16+16), all 64 columns ----
        v8f qk[4];
#pragma unroll
        for (int nt = 0; nt < 4; ++nt)
#pragma unroll
            for (int r = 0; r < 8; ++r) qk[nt][r] = 0.f;
#pragma unroll
        for (int ks = 0; ks < 64; ks += 32) {
            v16h a = lds_tile_16x32(&Qs[w * 16][ks], 72);
#pragma unroll
            for (int nt = 0; nt < 4; ++nt) {
                v16h b = lds_tile_16x32(&Ks[nt * 16][ks], 72);
                qk[nt] = wmma_f16(a, b, qk[nt]);
            }
        }

        // ---- scale + mask, online softmax (rows live in 16-lane halves) ----
        float sc[4][8];
#pragma unroll
        for (int nt = 0; nt < 4; ++nt)
#pragma unroll
            for (int r = 0; r < 8; ++r)
                sc[nt][r] = qk[nt][r] * 0.125f + mk[nt * 16 + nl] * (-1e12f);

        float ef[8];
#pragma unroll
        for (int r = 0; r < 8; ++r) {
            float mx = fmaxf(fmaxf(sc[0][r], sc[1][r]), fmaxf(sc[2][r], sc[3][r]));
#pragma unroll
            for (int off = 1; off < 16; off <<= 1)
                mx = fmaxf(mx, __shfl_xor(mx, off, 32));
            float mnew = fmaxf(mrow[r], mx);
            ef[r] = __expf(mrow[r] - mnew);
            float rs = 0.f;
#pragma unroll
            for (int nt = 0; nt < 4; ++nt) {
                float p = __expf(sc[nt][r] - mnew);
                sc[nt][r] = p;
                rs += p;
            }
#pragma unroll
            for (int off = 1; off < 16; off <<= 1) rs += __shfl_xor(rs, off, 32);
            lrow[r] = lrow[r] * ef[r] + rs;
            mrow[r] = mnew;
        }

        // rescale O, spill P (C layout) into per-wave LDS as f16
#pragma unroll
        for (int nt = 0; nt < 4; ++nt)
#pragma unroll
            for (int r = 0; r < 8; ++r) {
                o[nt][r] *= ef[r];
                Ps[w][r + 8 * half][nt * 16 + nl] = (_Float16)sc[nt][r];
            }
        __syncthreads();

        // ---- O += P V : A = P (16x64), B = Vt ----
#pragma unroll
        for (int ks = 0; ks < 64; ks += 32) {
            v16h a = lds_tile_16x32(&Ps[w][0][ks], 72);
#pragma unroll
            for (int nt = 0; nt < 4; ++nt) {
                v16h b = lds_tile_16x32(&Vt[nt * 16][ks], 72);
                o[nt] = wmma_f16(a, b, o[nt]);
            }
        }
        __syncthreads();
    }

    // ---- finalize: O / l, write ctx [B*S, D] f16 for the out-projection ----
#pragma unroll
    for (int r = 0; r < 8; ++r) {
        float inv = 1.f / lrow[r];
        int s = q0 + w * 16 + r + 8 * half;
#pragma unroll
        for (int nt = 0; nt < 4; ++nt) {
            int d = h * HDIM + nt * 16 + nl;
            ctx[((size_t)bb * SEQ + s) * DMODEL + d] = (_Float16)(o[nt][r] * inv);
        }
    }
}

// ---------------------------------------------------------------------------
extern "C" void kernel_launch(void* const* d_in, const int* in_sizes, int n_in,
                              void* d_out, int out_size, void* d_ws, size_t ws_size,
                              hipStream_t stream) {
    (void)in_sizes; (void)n_in; (void)out_size; (void)ws_size;

    const float* q    = (const float*)d_in[0];
    const float* k    = (const float*)d_in[1];
    const float* v    = (const float*)d_in[2];
    const float* mask = (const float*)d_in[3];
    const float* Wq   = (const float*)d_in[4];
    const float* bq   = (const float*)d_in[5];
    const float* Wk   = (const float*)d_in[6];
    const float* bk   = (const float*)d_in[7];
    const float* Wv   = (const float*)d_in[8];
    const float* bv   = (const float*)d_in[9];
    const float* Wo   = (const float*)d_in[10];
    const float* bo   = (const float*)d_in[11];
    float* out = (float*)d_out;

    // f16 workspace: qh | kh | vh | ctx  (4 x 8 MB = 32 MB)
    _Float16* ws  = (_Float16*)d_ws;
    const size_t per = (size_t)BATCH * HEADS * SEQ * HDIM;
    _Float16* qhh = ws;
    _Float16* khh = ws + per;
    _Float16* vhh = ws + 2 * per;
    _Float16* ctx = ws + 3 * per;

    dim3 gproj(DMODEL / 64, (BATCH * SEQ) / 128);  // 16 x 32 blocks

    gemm_proj_kernel<false, true><<<gproj, 256, 0, stream>>>(q, Wq, bq, qhh);
    gemm_proj_kernel<false, true><<<gproj, 256, 0, stream>>>(k, Wk, bk, khh);
    gemm_proj_kernel<false, true><<<gproj, 256, 0, stream>>>(v, Wv, bv, vhh);

    flash_attn_kernel<<<BATCH * HEADS * (SEQ / 128), 256, 0, stream>>>(
        qhh, khh, vhh, mask, ctx);

    gemm_proj_kernel<true, false><<<gproj, 256, 0, stream>>>(ctx, Wo, bo, out);
}
